// KeypointDetector_12601434046675
// MI455X (gfx1250) — matrix operand order repaired and checked by reading, not compiled, not run on hardware
//
#include <hip/hip_runtime.h>
#include <math.h>

typedef _Float16 h16;
typedef __attribute__((ext_vector_type(8)))  _Float16 v8h;
typedef __attribute__((ext_vector_type(16))) _Float16 v16h;
typedef __attribute__((ext_vector_type(8)))  float    v8f;

#define NPTS 40960
#define MA 256
#define MB 64

// Tensor Data Mover availability (device pass only; safe fallback otherwise)
#if defined(__AMDGCN__) && __has_builtin(__builtin_amdgcn_tensor_load_to_lds) && \
    __has_builtin(__builtin_amdgcn_s_wait_tensorcnt)
#define CDNA5_TDM 1
#else
#define CDNA5_TDM 0
#endif

#if CDNA5_TDM
typedef unsigned int u32x4 __attribute__((ext_vector_type(4)));
typedef int i32x8 __attribute__((ext_vector_type(8)));
typedef int i32x4 __attribute__((ext_vector_type(4)));
#endif

// ---------------------------------------------------------------------------
// WMMA fragment loaders (CDNA5 16-bit layouts, wave32)
// A 16x32 (MxK): lane<16 -> M=lane, V0..3: K-pairs 0..7, V4..7: K 16..23;
//                lane>=16 adds K+8.  => two contiguous 16B runs per lane.
// B 32x16 (KxN): lane<16 -> N=lane, same K layout as A: K 0..7, 16..23; lane>=16 adds K+8.
// ---------------------------------------------------------------------------
__device__ __forceinline__ v16h frag_a_ld(const h16* W, int m0, int k0, int ld) {
  int lane = threadIdx.x & 31;
  const h16* base = W + (size_t)(m0 + (lane & 15)) * (size_t)ld
                      + (size_t)(k0 + ((lane >> 4) << 3));
  v8h lo = *(const v8h*)(base);
  v8h hi = *(const v8h*)(base + 16);
  return __builtin_shufflevector(lo, hi, 0,1,2,3,4,5,6,7,8,9,10,11,12,13,14,15);
}

__device__ __forceinline__ v16h frag_b_ld(const h16* X, int n0, int k0, int ld) {
  int lane = threadIdx.x & 31;
  const h16* base = X + (size_t)(n0 + (lane & 15)) * (size_t)ld
                      + (size_t)(k0 + ((lane >> 4) << 3));
  v8h lo = *(const v8h*)(base);
  v8h hi = *(const v8h*)(base + 16);
  return __builtin_shufflevector(lo, hi, 0,1,2,3,4,5,6,7,8,9,10,11,12,13,14,15);
}

#define WMMA_F16(A, B, C) \
  __builtin_amdgcn_wmma_f32_16x16x32_f16(false, (A), false, (B), (short)0, (C), false, false)

// ---------------------------------------------------------------------------
// Generic small GEMM: out[n][coloff+m] = act(scale*sum_k W[m][k]*X[n][k] + bias[m])
// one wave per 16x16 output tile; M,Nn multiples of 16; K multiple of 32.
// ---------------------------------------------------------------------------
__global__ __launch_bounds__(256) void k_gemm(
    const h16* __restrict__ W, const h16* __restrict__ X,
    const float* __restrict__ bias, h16* __restrict__ out,
    int M, int K, int Nn, int ldx, int ldo, int coloff, int relu, float scale)
{
  int wave = threadIdx.x >> 5;
  int tile = blockIdx.x * 8 + wave;
  int mtiles = M >> 4;
  int tiles  = mtiles * (Nn >> 4);
  if (tile >= tiles) return;
  int m0 = (tile % mtiles) << 4;
  int n0 = (tile / mtiles) << 4;
  v8f acc = {0.f,0.f,0.f,0.f,0.f,0.f,0.f,0.f};
  for (int k0 = 0; k0 < K; k0 += 32) {
    v16h a  = frag_a_ld(W, m0, k0, K);
    v16h bb = frag_b_ld(X, n0, k0, ldx);
    acc = WMMA_F16(a, bb, acc);
  }
  int lane = threadIdx.x & 31;
  int hh = lane >> 4, nn = lane & 15;
  size_t row = (size_t)(n0 + nn) * (size_t)ldo + (size_t)coloff;
#pragma unroll
  for (int r = 0; r < 8; ++r) {
    int m = m0 + r + hh * 8;
    float v = acc[r] * scale;
    if (bias) v += bias[m];
    if (relu) v = fmaxf(v, 0.f);
    out[row + m] = (h16)v;
  }
}

// ---------------------------------------------------------------------------
// Packing helpers
// ---------------------------------------------------------------------------
__global__ void k_cvt(const float* __restrict__ src, h16* __restrict__ dst, int n) {
  int i = blockIdx.x * blockDim.x + threadIdx.x;
  if (i < n) dst[i] = (h16)src[i];
}
__global__ void k_cvt_pad(const float* __restrict__ src, h16* __restrict__ dst,
                          int rows, int K, int ldd) {
  int e = blockIdx.x * blockDim.x + threadIdx.x;
  if (e >= rows * K) return;
  int r = e / K, c = e % K;
  dst[(size_t)r * ldd + c] = (h16)src[e];
}
__global__ void k_copy_f32(const float* __restrict__ src, float* __restrict__ dst, int n) {
  int i = blockIdx.x * blockDim.x + threadIdx.x;
  if (i < n) dst[i] = src[i];
}
// channel-major f32 [C][P] -> point-major f16 [P][ldo] at column offset
__global__ void k_transpose_f16(const float* __restrict__ src, h16* __restrict__ dst,
                                int C, int P, int ldo, int coloff) {
  int e = blockIdx.x * blockDim.x + threadIdx.x;
  if (e >= C * P) return;
  int c = e % C, p = e / C;
  dst[(size_t)p * ldo + coloff + c] = (h16)src[(size_t)c * P + p];
}
// broadcast vector v[C] into every point row
__global__ void k_bcast_f16(const float* __restrict__ v, h16* __restrict__ dst,
                            int C, int P, int ldo, int coloff) {
  int e = blockIdx.x * blockDim.x + threadIdx.x;
  if (e >= C * P) return;
  int c = e % C, p = e / C;
  dst[(size_t)p * ldo + coloff + c] = (h16)v[c];
}

// ---------------------------------------------------------------------------
// top-3 (smallest) tracker
// ---------------------------------------------------------------------------
struct Top3 { float d0, d1, d2; int i0, i1, i2; };
__device__ __forceinline__ void top3_insert(Top3& t, float d, int j) {
  if (d < t.d0)      { t.d2=t.d1; t.i2=t.i1; t.d1=t.d0; t.i1=t.i0; t.d0=d; t.i0=j; }
  else if (d < t.d1) { t.d2=t.d1; t.i2=t.i1; t.d1=d; t.i1=j; }
  else if (d < t.d2) { t.d2=d; t.i2=j; }
}

// ---------------------------------------------------------------------------
// interp_ab: 256 a-nodes -> top3 of 64 b-nodes, gather up_b (512ch) into Xa cols 64..575
// ---------------------------------------------------------------------------
__global__ __launch_bounds__(256) void k_interp_ab(
    const float* __restrict__ na, const float* __restrict__ nb,
    const h16* __restrict__ upbT, h16* __restrict__ Xa)
{
  int p = threadIdx.x;
  float ax = na[p], ay = na[256 + p], az = na[512 + p];
  Top3 t = {1e30f, 1e30f, 1e30f, 0, 0, 0};
  for (int j = 0; j < 64; ++j) {
    float dx = ax - nb[j], dy = ay - nb[64 + j], dz = az - nb[128 + j];
    top3_insert(t, dx*dx + dy*dy + dz*dz, j);
  }
  float e0 = sqrtf(t.d0), e1 = sqrtf(t.d1), e2 = sqrtf(t.d2);
  float s = e0 + e1 + e2; s = (s > 0.f) ? s : 1.f;
  float w0 = 1.f - e0 / s, w1 = 1.f - e1 / s, w2 = 1.f - e2 / s;
  const h16* c0 = upbT + (size_t)t.i0 * 512;
  const h16* c1 = upbT + (size_t)t.i1 * 512;
  const h16* c2 = upbT + (size_t)t.i2 * 512;
  for (int c = 0; c < 512; ++c) {
    float v = w0 * (float)c0[c] + w1 * (float)c1[c] + w2 * (float)c2[c];
    Xa[(size_t)p * 832 + 64 + c] = (h16)v;
  }
}

// ---------------------------------------------------------------------------
// Fused per-point kernel: interp_pa + interp_pb + pn_point MLP [736,256,256,82]
// 64-point tiles, 256 threads (8 waves). Dynamic LDS: X[64][736] f16,
// H1[64][256] f16, H2 aliases X after layer 1.  TDM stages first/second_pn_out
// tiles into static LDS while interp runs.  Rows 0..1 -> coarse, 2..81 -> fine.
// ---------------------------------------------------------------------------
#define L1K 736
#define HCH 256
#define LDS_HALFS (64 * L1K + 64 * HCH)

#if CDNA5_TDM
// Issue a 2-D TDM tile load: tile_x elems (4B) x tile_y rows, row stride = stride_x
__device__ __forceinline__ void tdm_load_2d(const void* gaddr, void* ldsaddr,
                                            unsigned tile_x, unsigned tile_y,
                                            unsigned tensor_x, unsigned tensor_y,
                                            unsigned stride_x) {
  unsigned long long ga = (unsigned long long)(size_t)gaddr;
  unsigned la = (unsigned)(unsigned long long)(size_t)ldsaddr; // low 32b = LDS offset
  u32x4 g0;
  g0[0] = 1u;                                            // count=1 (valid user D#)
  g0[1] = la;                                            // lds_addr
  g0[2] = (unsigned)(ga & 0xFFFFFFFFu);                  // global_addr[31:0]
  g0[3] = (unsigned)((ga >> 32) & 0x1FFFFFFu) | (2u << 30); // addr[56:32] | type=2
  i32x8 g1;
  g1[0] = (int)(2u << 16);                               // workgroup_mask=0, data_size=4B
  g1[1] = (int)((tensor_x & 0xFFFFu) << 16);             // tensor_dim0[15:0]
  g1[2] = (int)((tensor_x >> 16) | ((tensor_y & 0xFFFFu) << 16));
  g1[3] = (int)((tensor_y >> 16) | (tile_x << 16));      // tile_dim0
  g1[4] = (int)tile_y;                                   // tile_dim1, tile_dim2=0
  g1[5] = (int)stride_x;                                 // tensor_dim0_stride[31:0]
  g1[6] = 0;
  g1[7] = 0;
  i32x4 z4 = {0, 0, 0, 0};
#if __clang_major__ >= 23
  i32x8 z8 = {0, 0, 0, 0, 0, 0, 0, 0};
  __builtin_amdgcn_tensor_load_to_lds(g0, g1, z4, z4, z8, 0);
#else
  __builtin_amdgcn_tensor_load_to_lds(g0, g1, z4, z4, 0);
#endif
}
#endif

__global__ __launch_bounds__(256) void k_point(
    const float* __restrict__ pc, const float* __restrict__ node_a,
    const float* __restrict__ node_b, const int* __restrict__ idx_a,
    const float* __restrict__ fpn, const float* __restrict__ spn,
    const h16* __restrict__ upbT0, const h16* __restrict__ upbT1,
    const h16* __restrict__ upaT0, const h16* __restrict__ upaT1,
    const h16* __restrict__ W1, const float* __restrict__ bias1,
    const h16* __restrict__ W2, const float* __restrict__ bias2,
    const h16* __restrict__ W3, const float* __restrict__ bias3,
    float* __restrict__ out)
{
  extern __shared__ h16 lds[];
  h16* Xl = lds;                 // [64][736]
  h16* H1 = lds + 64 * L1K;      // [64][256]
  h16* H2 = lds;                 // alias Xl once layer 1 is done

  __shared__ float wpb[64][3]; __shared__ int ipb[64][3];
  __shared__ float wpa[64][3]; __shared__ int ipa[64][3];
#if CDNA5_TDM
  __shared__ float Sfp[32 * 64]; // staged first_pn_out tile  [c][p]
  __shared__ float Ssp[64 * 64]; // staged second_pn_out tile [c][p]
#endif

  const int b   = blockIdx.y;
  const int n0  = blockIdx.x * 64;
  const int tid = threadIdx.x;
  const int N   = NPTS;
  const float* pcb = pc + (size_t)b * 3 * N;
  const float* fb  = fpn + (size_t)b * 32 * N;
  const float* sb  = spn + (size_t)b * 64 * N;

#if CDNA5_TDM
  // kick off the DMA of the strided 2-D activation tiles; overlaps with interp
  if (tid < 32) {
    tdm_load_2d(fb + n0, Sfp, 64, 32, (unsigned)N, 32, (unsigned)N);
    tdm_load_2d(sb + n0, Ssp, 64, 64, (unsigned)N, 64, (unsigned)N);
  }
#endif

  // ---- phase 1a: per-point neighbor indices + weights (waves 0..3) ----
  if (tid < 64) {                      // top-3 vs node_b (waves 0-1, uniform)
    int p = tid, n = n0 + p;
    float ax = pcb[n], ay = pcb[N + n], az = pcb[2 * N + n];
    const float* nb = node_b + (size_t)b * 3 * MB;
    Top3 t = {1e30f, 1e30f, 1e30f, 0, 0, 0};
    for (int j = 0; j < 64; ++j) {
      float dx = ax - nb[j], dy = ay - nb[64 + j], dz = az - nb[128 + j];
      top3_insert(t, dx*dx + dy*dy + dz*dz, j);
    }
    float e0 = sqrtf(t.d0), e1 = sqrtf(t.d1), e2 = sqrtf(t.d2);
    float s = e0 + e1 + e2; s = (s > 0.f) ? s : 1.f;
    wpb[p][0] = 1.f - e0 / s; wpb[p][1] = 1.f - e1 / s; wpb[p][2] = 1.f - e2 / s;
    ipb[p][0] = t.i0; ipb[p][1] = t.i1; ipb[p][2] = t.i2;
  } else if (tid < 128) {              // given indices vs node_a (waves 2-3)
    int p = tid - 64, n = n0 + p;
    float ax = pcb[n], ay = pcb[N + n], az = pcb[2 * N + n];
    const float* na = node_a + (size_t)b * 3 * MA;
    const int* ia = idx_a + ((size_t)b * N + n) * 3;
    float e[3]; int id[3];
#pragma unroll
    for (int i = 0; i < 3; ++i) {
      int j = ia[i]; id[i] = j;
      float dx = ax - na[j], dy = ay - na[256 + j], dz = az - na[512 + j];
      e[i] = sqrtf(dx*dx + dy*dy + dz*dz);
    }
    float s = e[0] + e[1] + e[2]; s = (s > 0.f) ? s : 1.f;
#pragma unroll
    for (int i = 0; i < 3; ++i) { wpa[p][i] = 1.f - e[i] / s; ipa[p][i] = id[i]; }
  }
  __syncthreads();

  // ---- phase 1b: assemble X tile (f16) ----
  const h16* ub = (b == 0) ? upbT0 : upbT1;   // [64][512]
  const h16* ua = (b == 0) ? upaT0 : upaT1;   // [256][128]
  for (int e = tid; e < 64 * 512; e += 256) { // interp_pb -> cols 128..639
    int p = e >> 9, c = e & 511;
    float v = wpb[p][0] * (float)ub[(size_t)ipb[p][0] * 512 + c]
            + wpb[p][1] * (float)ub[(size_t)ipb[p][1] * 512 + c]
            + wpb[p][2] * (float)ub[(size_t)ipb[p][2] * 512 + c];
    Xl[(size_t)p * L1K + 128 + c] = (h16)v;
  }
  for (int e = tid; e < 64 * 128; e += 256) { // interp_pa -> cols 0..127
    int p = e >> 7, c = e & 127;
    float v = wpa[p][0] * (float)ua[(size_t)ipa[p][0] * 128 + c]
            + wpa[p][1] * (float)ua[(size_t)ipa[p][1] * 128 + c]
            + wpa[p][2] * (float)ua[(size_t)ipa[p][2] * 128 + c];
    Xl[(size_t)p * L1K + c] = (h16)v;
  }
#if CDNA5_TDM
  __builtin_amdgcn_s_wait_tensorcnt(0);       // no-op for waves that issued nothing
  __syncthreads();
  for (int e = tid; e < 64 * 32; e += 256) {  // first_pn_out -> cols 640..671
    int p = e & 63, c = e >> 6;
    Xl[(size_t)p * L1K + 640 + c] = (h16)Sfp[c * 64 + p];
  }
  for (int e = tid; e < 64 * 64; e += 256) {  // second_pn_out -> cols 672..735
    int p = e & 63, c = e >> 6;
    Xl[(size_t)p * L1K + 672 + c] = (h16)Ssp[c * 64 + p];
  }
#else
  for (int e = tid; e < 64 * 32; e += 256) {  // first_pn_out -> cols 640..671
    int p = e & 63, c = e >> 6;
    Xl[(size_t)p * L1K + 640 + c] = (h16)fb[(size_t)c * N + n0 + p];
  }
  for (int e = tid; e < 64 * 64; e += 256) {  // second_pn_out -> cols 672..735
    int p = e & 63, c = e >> 6;
    Xl[(size_t)p * L1K + 672 + c] = (h16)sb[(size_t)c * N + n0 + p];
  }
#endif
  __syncthreads();

  const int wave = tid >> 5;
  const int lane = tid & 31, hh = lane >> 4, nn = lane & 15;
  const v8f vzero = {0.f,0.f,0.f,0.f,0.f,0.f,0.f,0.f};

  // ---- layer 1: 256x736 @ 736x64, ReLU -> H1 ----
  {
    v8f acc[2][4];
#pragma unroll
    for (int mi = 0; mi < 2; ++mi)
#pragma unroll
      for (int nt = 0; nt < 4; ++nt) acc[mi][nt] = vzero;
    int mw = wave * 32;
    for (int k0 = 0; k0 < L1K; k0 += 32) {
      if (k0 + 32 < L1K)
        __builtin_prefetch(W1 + (size_t)mw * L1K + k0 + 32, 0, 1);
      v16h a0 = frag_a_ld(W1, mw,      k0, L1K);
      v16h a1 = frag_a_ld(W1, mw + 16, k0, L1K);
#pragma unroll
      for (int nt = 0; nt < 4; ++nt) {
        v16h bb = frag_b_ld(Xl, nt * 16, k0, L1K);
        acc[0][nt] = WMMA_F16(a0, bb, acc[0][nt]);
        acc[1][nt] = WMMA_F16(a1, bb, acc[1][nt]);
      }
    }
#pragma unroll
    for (int mi = 0; mi < 2; ++mi)
#pragma unroll
      for (int nt = 0; nt < 4; ++nt)
#pragma unroll
        for (int r = 0; r < 8; ++r) {
          int m = mw + mi * 16 + r + hh * 8;
          float v = fmaxf(acc[mi][nt][r] + bias1[m], 0.f);
          H1[(size_t)(nt * 16 + nn) * HCH + m] = (h16)v;
        }
  }
  __syncthreads();

  // ---- layer 2: 256x256 @ 256x64, ReLU -> H2 (aliases X) ----
  {
    v8f acc[2][4];
#pragma unroll
    for (int mi = 0; mi < 2; ++mi)
#pragma unroll
      for (int nt = 0; nt < 4; ++nt) acc[mi][nt] = vzero;
    int mw = wave * 32;
    for (int k0 = 0; k0 < 256; k0 += 32) {
      v16h a0 = frag_a_ld(W2, mw,      k0, 256);
      v16h a1 = frag_a_ld(W2, mw + 16, k0, 256);
#pragma unroll
      for (int nt = 0; nt < 4; ++nt) {
        v16h bb = frag_b_ld(H1, nt * 16, k0, HCH);
        acc[0][nt] = WMMA_F16(a0, bb, acc[0][nt]);
        acc[1][nt] = WMMA_F16(a1, bb, acc[1][nt]);
      }
    }
    __syncthreads();
#pragma unroll
    for (int mi = 0; mi < 2; ++mi)
#pragma unroll
      for (int nt = 0; nt < 4; ++nt)
#pragma unroll
        for (int r = 0; r < 8; ++r) {
          int m = mw + mi * 16 + r + hh * 8;
          float v = fmaxf(acc[mi][nt][r] + bias2[m], 0.f);
          H2[(size_t)(nt * 16 + nn) * HCH + m] = (h16)v;
        }
  }
  __syncthreads();

  // ---- layer 3: 96x256 @ 256x64 (rows 82..95 zero pad) -> d_out ----
  for (int i = 0; i < 3; ++i) {
    int t = wave * 3 + i;          // 24 tiles: 6 m-tiles x 4 n-tiles
    int mt = t % 6, nt = t / 6;
    v8f acc = vzero;
    for (int k0 = 0; k0 < 256; k0 += 32) {
      v16h a  = frag_a_ld(W3, mt * 16, k0, 256);
      v16h bb = frag_b_ld(H2, nt * 16, k0, HCH);
      acc = WMMA_F16(a, bb, acc);
    }
#pragma unroll
    for (int r = 0; r < 8; ++r) {
      int m = mt * 16 + r + hh * 8;
      float v = acc[r] + bias3[m];
      int n = n0 + nt * 16 + nn;
      if (m < 2)        out[((size_t)b * 2 + m) * (size_t)N + n] = v;              // coarse
      else if (m < 82)  out[(size_t)4 * N + ((size_t)b * 80 + (m - 2)) * (size_t)N + n] = v; // fine
    }
  }
}

// ---------------------------------------------------------------------------
// Host: full pipeline
// ---------------------------------------------------------------------------
extern "C" void kernel_launch(void* const* d_in, const int* in_sizes, int n_in,
                              void* d_out, int out_size, void* d_ws, size_t ws_size,
                              hipStream_t stream) {
  (void)in_sizes; (void)n_in; (void)out_size; (void)ws_size;
  const float* pc    = (const float*)d_in[0];
  const float* na_p  = (const float*)d_in[1];
  const float* nb_p  = (const float*)d_in[2];
  const int*   idx_a = (const int*)  d_in[3];
  const float* fpn   = (const float*)d_in[4];
  const float* spn   = (const float*)d_in[5];
  const float* naf   = (const float*)d_in[6];
  const float* nbf   = (const float*)d_in[7];
  const float* gfeat = (const float*)d_in[8];
  const float* s16   = (const float*)d_in[9];
  const float* s32   = (const float*)d_in[10];
  const float* gimg  = (const float*)d_in[11];
  float* out = (float*)d_out;

  size_t cur = 0;
  auto alloc = [&](size_t elems, size_t esz) -> char* {
    cur = (cur + 255) & ~(size_t)255;
    char* p = (char*)d_ws + cur;
    cur += elems * esz;
    return p;
  };
  auto cdiv = [](int a, int b) { return (a + b - 1) / b; };
  auto cvt = [&](int which, h16* dst, int n) {
    k_cvt<<<cdiv(n, 256), 256, 0, stream>>>((const float*)d_in[which], dst, n);
  };

  // f16 weights
  h16* wBattW1 = (h16*)alloc(256 * 768, 2);
  h16* wBattW2 = (h16*)alloc(80 * 256, 2);
  h16* wBW1 = (h16*)alloc(1024 * 1792, 2);
  h16* wBW2 = (h16*)alloc(512 * 1024, 2);
  h16* wBW3 = (h16*)alloc(512 * 512, 2);
  h16* wAattW1 = (h16*)alloc(256 * 576, 2);
  h16* wAattW2 = (h16*)alloc(320 * 256, 2);
  h16* wAW1 = (h16*)alloc(512 * 832, 2);
  h16* wAW2 = (h16*)alloc(128 * 512, 2);
  h16* wAW3 = (h16*)alloc(128 * 128, 2);
  h16* wPW1 = (h16*)alloc(256 * 736, 2);
  h16* wPW2 = (h16*)alloc(256 * 256, 2);
  h16* wPW3 = (h16*)alloc(96 * 256, 2);         // padded 82 -> 96 rows
  float* pb3pad = (float*)alloc(96, 4);
  h16* s16h = (h16*)alloc(2 * 256 * 320, 2);    // [b][256][320]
  h16* s32h = (h16*)alloc((size_t)2 * 512 * 96, 2); // [b][512][96] K padded 80->96

  h16 *XbAtt[2], *HbAtt[2], *AttB[2], *Xb[2], *Hb1[2], *Hb2[2], *UpbT[2];
  h16 *XaAtt[2], *HaAtt[2], *AttA[2], *Xa[2], *Ha1[2], *Ha2[2], *UpaT[2];
  for (int b = 0; b < 2; ++b) {
    XbAtt[b] = (h16*)alloc(64 * 768, 2);
    HbAtt[b] = (h16*)alloc(64 * 256, 2);
    AttB[b]  = (h16*)alloc(64 * 96, 2);
    Xb[b]    = (h16*)alloc(64 * 1792, 2);
    Hb1[b]   = (h16*)alloc(64 * 1024, 2);
    Hb2[b]   = (h16*)alloc(64 * 512, 2);
    UpbT[b]  = (h16*)alloc(64 * 512, 2);
    XaAtt[b] = (h16*)alloc(256 * 576, 2);
    HaAtt[b] = (h16*)alloc(256 * 256, 2);
    AttA[b]  = (h16*)alloc(256 * 320, 2);
    Xa[b]    = (h16*)alloc(256 * 832, 2);
    Ha1[b]   = (h16*)alloc(256 * 512, 2);
    Ha2[b]   = (h16*)alloc(256 * 128, 2);
    UpaT[b]  = (h16*)alloc(256 * 128, 2);
  }

  // zero the buffers that carry zero-padding
  hipMemsetAsync(wPW3, 0, (size_t)96 * 256 * 2, stream);
  hipMemsetAsync(pb3pad, 0, 96 * 4, stream);
  hipMemsetAsync(s32h, 0, (size_t)2 * 512 * 96 * 2, stream);
  hipMemsetAsync(AttB[0], 0, 64 * 96 * 2, stream);
  hipMemsetAsync(AttB[1], 0, 64 * 96 * 2, stream);

  // weight conversion (f32 -> f16)
  cvt(12, wBattW1, 256 * 768);  cvt(14, wBattW2, 80 * 256);
  cvt(16, wBW1, 1024 * 1792);   cvt(18, wBW2, 512 * 1024);  cvt(20, wBW3, 512 * 512);
  cvt(22, wAattW1, 256 * 576);  cvt(24, wAattW2, 320 * 256);
  cvt(26, wAW1, 512 * 832);     cvt(28, wAW2, 128 * 512);   cvt(30, wAW3, 128 * 128);
  cvt(32, wPW1, 256 * 736);     cvt(34, wPW2, 256 * 256);
  cvt(36, wPW3, 82 * 256);      // rows 0..81; rest stays zero
  k_copy_f32<<<1, 96, 0, stream>>>((const float*)d_in[37], pb3pad, 82);
  k_cvt<<<cdiv(2 * 256 * 320, 256), 256, 0, stream>>>(s16, s16h, 2 * 256 * 320);
  for (int b = 0; b < 2; ++b)
    k_cvt_pad<<<cdiv(512 * 80, 256), 256, 0, stream>>>(
        s32 + (size_t)b * 512 * 80, s32h + (size_t)b * 512 * 96, 512, 80, 96);

  const float* battB1 = (const float*)d_in[13];
  const float* battB2 = (const float*)d_in[15];
  const float* bB1 = (const float*)d_in[17];
  const float* bB2 = (const float*)d_in[19];
  const float* bB3 = (const float*)d_in[21];
  const float* aattB1 = (const float*)d_in[23];
  const float* aattB2 = (const float*)d_in[25];
  const float* aB1 = (const float*)d_in[27];
  const float* aB2 = (const float*)d_in[29];
  const float* aB3 = (const float*)d_in[31];
  const float* pB1 = (const float*)d_in[33];
  const float* pB2 = (const float*)d_in[35];

  auto gemm = [&](const h16* W, const h16* X, const float* bias, h16* o,
                  int M, int K, int Nn, int ldx, int ldo, int coloff,
                  int relu, float scale) {
    int tiles = (M / 16) * (Nn / 16);
    k_gemm<<<cdiv(tiles, 8), 256, 0, stream>>>(W, X, bias, o, M, K, Nn, ldx, ldo,
                                               coloff, relu, scale);
  };

  for (int b = 0; b < 2; ++b) {
    // --- pack node-level concat inputs ---
    k_transpose_f16<<<cdiv(256 * 64, 256), 256, 0, stream>>>(nbf + (size_t)b * 256 * 64, XbAtt[b], 256, 64, 768, 0);
    k_bcast_f16<<<cdiv(512 * 64, 256), 256, 0, stream>>>(gimg + (size_t)b * 512, XbAtt[b], 512, 64, 768, 256);
    k_transpose_f16<<<cdiv(256 * 64, 256), 256, 0, stream>>>(nbf + (size_t)b * 256 * 64, Xb[b], 256, 64, 1792, 0);
    k_bcast_f16<<<cdiv(512 * 64, 256), 256, 0, stream>>>(gfeat + (size_t)b * 512, Xb[b], 512, 64, 1792, 256);
    k_bcast_f16<<<cdiv(512 * 64, 256), 256, 0, stream>>>(gimg + (size_t)b * 512, Xb[b], 512, 64, 1792, 1280);
    k_transpose_f16<<<cdiv(64 * 256, 256), 256, 0, stream>>>(naf + (size_t)b * 64 * 256, XaAtt[b], 64, 256, 576, 0);
    k_bcast_f16<<<cdiv(512 * 256, 256), 256, 0, stream>>>(gimg + (size_t)b * 512, XaAtt[b], 512, 256, 576, 64);
    k_transpose_f16<<<cdiv(64 * 256, 256), 256, 0, stream>>>(naf + (size_t)b * 64 * 256, Xa[b], 64, 256, 832, 0);

    // --- b-branch: attention, image weighting, pn_b ---
    gemm(wBattW1, XbAtt[b], battB1, HbAtt[b], 256, 768, 64, 768, 256, 0, 1, 1.f);
    gemm(wBattW2, HbAtt[b], battB2, AttB[b],  80,  256, 64, 256, 96,  0, 0, 1.f);
    gemm(s32h + (size_t)b * 512 * 96, AttB[b], nullptr, Xb[b],
         512, 96, 64, 96, 1792, 768, 0, 1.f / 80.f);
    gemm(wBW1, Xb[b],  bB1, Hb1[b],  1024, 1792, 64, 1792, 1024, 0, 1, 1.f);
    gemm(wBW2, Hb1[b], bB2, Hb2[b],  512,  1024, 64, 1024, 512,  0, 1, 1.f);
    gemm(wBW3, Hb2[b], bB3, UpbT[b], 512,  512,  64, 512,  512,  0, 0, 1.f);

    // --- a-branch: attention, image weighting, interp_ab, pn_a ---
    gemm(wAattW1, XaAtt[b], aattB1, HaAtt[b], 256, 576, 256, 576, 256, 0, 1, 1.f);
    gemm(wAattW2, HaAtt[b], aattB2, AttA[b],  320, 256, 256, 256, 320, 0, 0, 1.f);
    gemm(s16h + (size_t)b * 256 * 320, AttA[b], nullptr, Xa[b],
         256, 320, 256, 320, 832, 576, 0, 1.f / 320.f);
    k_interp_ab<<<1, 256, 0, stream>>>(na_p + (size_t)b * 3 * MA,
                                       nb_p + (size_t)b * 3 * MB, UpbT[b], Xa[b]);
    gemm(wAW1, Xa[b],  aB1, Ha1[b],  512, 832, 256, 832, 512, 0, 1, 1.f);
    gemm(wAW2, Ha1[b], aB2, Ha2[b],  128, 512, 256, 512, 128, 0, 1, 1.f);
    gemm(wAW3, Ha2[b], aB3, UpaT[b], 128, 128, 256, 128, 128, 0, 0, 1.f);
  }

  // --- fused per-point stage ---
  const int ldsBytes = LDS_HALFS * 2; // 126976 B dynamic (<320 KB/WGP incl. static)
  hipFuncSetAttribute((const void*)k_point,
                      hipFuncAttributeMaxDynamicSharedMemorySize, ldsBytes);
  dim3 grid(NPTS / 64, 2);
  k_point<<<grid, 256, ldsBytes, stream>>>(
      pc, na_p, nb_p, idx_a, fpn, spn,
      UpbT[0], UpbT[1], UpaT[0], UpaT[1],
      wPW1, pB1, wPW2, pB2, wPW3, pb3pad, out);
}
